// ProtypicalNet_89026082111519
// MI455X (gfx1250) — matrix operand order repaired and checked by reading, not compiled, not run on hardware
//
#include <hip/hip_runtime.h>

typedef __bf16 bf16_t;
typedef __attribute__((ext_vector_type(16))) __bf16 v16bf;
typedef __attribute__((ext_vector_type(8)))  __bf16 v8bf;
typedef __attribute__((ext_vector_type(8)))  float   v8f;

constexpr int CLS   = 1024;   // K (prototypes)
constexpr int DIM   = 512;    // D
constexpr int BATCH = 16384;  // B
constexpr float INV_TAU = 10.0f;

// ---------------------------------------------------------------------------
// zero the sim_loss accumulator (graph replays must be deterministic)
// ---------------------------------------------------------------------------
__global__ void zero_kernel(float* acc) {
    if (threadIdx.x == 0) acc[0] = 0.0f;
}

// ---------------------------------------------------------------------------
// x = l2norm(input) rows -> bf16 workspace
// ---------------------------------------------------------------------------
__global__ __launch_bounds__(256) void norm_x_kernel(const float* __restrict__ in,
                                                     bf16_t* __restrict__ outb) {
    const int row = blockIdx.x;
    const int t   = threadIdx.x;
    const float* r = in + (size_t)row * DIM;
    float a0 = r[t];
    float a1 = r[t + 256];
    __shared__ float red[256];
    red[t] = a0 * a0 + a1 * a1;
    __syncthreads();
    for (int o = 128; o > 0; o >>= 1) {
        if (t < o) red[t] += red[t + o];
        __syncthreads();
    }
    float inv = 1.0f / fmaxf(sqrtf(red[0]), 1e-12f);
    bf16_t* o = outb + (size_t)row * DIM;
    o[t]       = (bf16_t)(a0 * inv);
    o[t + 256] = (bf16_t)(a1 * inv);
}

// ---------------------------------------------------------------------------
// p = l2norm(prototypes) -> f32 output, bf16 copy, bf16 transpose [DIM][CLS]
// ---------------------------------------------------------------------------
__global__ __launch_bounds__(256) void norm_p_kernel(const float* __restrict__ in,
                                                     float*  __restrict__ pf32,
                                                     bf16_t* __restrict__ pb,
                                                     bf16_t* __restrict__ pT) {
    const int row = blockIdx.x;
    const int t   = threadIdx.x;
    const float* r = in + (size_t)row * DIM;
    float a0 = r[t];
    float a1 = r[t + 256];
    __shared__ float red[256];
    red[t] = a0 * a0 + a1 * a1;
    __syncthreads();
    for (int o = 128; o > 0; o >>= 1) {
        if (t < o) red[t] += red[t + o];
        __syncthreads();
    }
    float inv = 1.0f / fmaxf(sqrtf(red[0]), 1e-12f);
    float v0 = a0 * inv, v1 = a1 * inv;
    pf32[(size_t)row * DIM + t]         = v0;
    pf32[(size_t)row * DIM + t + 256]   = v1;
    pb[(size_t)row * DIM + t]           = (bf16_t)v0;
    pb[(size_t)row * DIM + t + 256]     = (bf16_t)v1;
    pT[(size_t)t * CLS + row]           = (bf16_t)v0;
    pT[(size_t)(t + 256) * CLS + row]   = (bf16_t)v1;
}

// ---------------------------------------------------------------------------
// Fused: sim tile (16 x 1024) via bf16 WMMA into LDS, then gumbel-softmax,
// write e (bf16) + accumulate sum_k e*sim for sim_loss. One wave per block.
// 4 independent WMMA chains per A-fragment for latency tolerance.
// ---------------------------------------------------------------------------
__global__ __launch_bounds__(32) void sim_softmax_kernel(const bf16_t* __restrict__ xb,
                                                         const bf16_t* __restrict__ pb,
                                                         const float*  __restrict__ gum,
                                                         bf16_t* __restrict__ eb,
                                                         float*  __restrict__ accDot) {
    extern __shared__ float simLds[];                 // [16][1028] padded
    const int lane = threadIdx.x;
    const int half = lane >> 4;                       // 0 or 1 (K-half select)
    const int l16  = lane & 15;
    const int m0   = blockIdx.x * 16;
    const int aRow = m0 + l16;

    // Hoist all A fragments for this 16-row tile (K = 512 -> 16 steps of 32)
    v16bf aF[16];
#pragma unroll
    for (int ks = 0; ks < 16; ++ks) {
        const bf16_t* base = xb + (size_t)aRow * DIM + ks * 32 + half * 16;
        v8bf lo = *(const v8bf*)(base);
        v8bf hi = *(const v8bf*)(base + 8);
#pragma unroll
        for (int i = 0; i < 8; ++i) { aF[ks][i] = lo[i]; aF[ks][i + 8] = hi[i]; }
    }

    // 64 n-tiles in groups of 4: four independent WMMA accumulation chains
    for (int nt = 0; nt < 64; nt += 4) {
        const int n0 = nt * 16;
        v8f acc0 = {}, acc1 = {}, acc2 = {}, acc3 = {};
        const bf16_t* b0base = pb + (size_t)(n0 + l16) * DIM + half * 16;
        const bf16_t* b1base = b0base + (size_t)16 * DIM;
        const bf16_t* b2base = b0base + (size_t)32 * DIM;
        const bf16_t* b3base = b0base + (size_t)48 * DIM;
#pragma unroll
        for (int ks = 0; ks < 16; ++ks) {
            v16bf b0 = *(const v16bf*)(b0base + ks * 32);
            v16bf b1 = *(const v16bf*)(b1base + ks * 32);
            v16bf b2 = *(const v16bf*)(b2base + ks * 32);
            v16bf b3 = *(const v16bf*)(b3base + ks * 32);
            acc0 = __builtin_amdgcn_wmma_f32_16x16x32_bf16(false, aF[ks], false, b0,
                                                           (short)0, acc0, false, false);
            acc1 = __builtin_amdgcn_wmma_f32_16x16x32_bf16(false, aF[ks], false, b1,
                                                           (short)0, acc1, false, false);
            acc2 = __builtin_amdgcn_wmma_f32_16x16x32_bf16(false, aF[ks], false, b2,
                                                           (short)0, acc2, false, false);
            acc3 = __builtin_amdgcn_wmma_f32_16x16x32_bf16(false, aF[ks], false, b3,
                                                           (short)0, acc3, false, false);
        }
#pragma unroll
        for (int r = 0; r < 8; ++r) {
            float* dst = simLds + (half * 8 + r) * 1028 + n0 + l16;
            dst[0]  = acc0[r];
            dst[16] = acc1[r];
            dst[32] = acc2[r];
            dst[48] = acc3[r];
        }
    }
    __syncthreads();   // single-wave WG: cheap; orders LDS writes vs reads

    // Gumbel-softmax per row; fuse sum_k e*sim
    float rowDotSum = 0.0f;
    for (int r = 0; r < 16; ++r) {
        const float* srow = simLds + r * 1028;
        const float* grow = gum + (size_t)(m0 + r) * CLS;
        float sv[32], lv[32];
        float mx = -3.4e38f;
#pragma unroll
        for (int j = 0; j < 32; ++j) {
            const int c = lane + j * 32;
            float u = grow[c];
            float g = -__logf(-__logf(u));
            sv[j] = srow[c];
            lv[j] = (sv[j] + g) * INV_TAU;
            mx = fmaxf(mx, lv[j]);
        }
        for (int o = 16; o > 0; o >>= 1) mx = fmaxf(mx, __shfl_xor(mx, o, 32));
        float Z = 0.0f, Dd = 0.0f;
#pragma unroll
        for (int j = 0; j < 32; ++j) {
            float t = __expf(lv[j] - mx);
            Z  += t;
            Dd += t * sv[j];
            lv[j] = t;
        }
        for (int o = 16; o > 0; o >>= 1) {
            Z  += __shfl_xor(Z, o, 32);
            Dd += __shfl_xor(Dd, o, 32);
        }
        const float invZ = 1.0f / Z;
        bf16_t* erow = eb + (size_t)(m0 + r) * CLS;
#pragma unroll
        for (int j = 0; j < 32; ++j) {
            erow[lane + j * 32] = (bf16_t)(lv[j] * invZ);
        }
        rowDotSum += Dd * invZ;
    }
    if (lane == 0) atomicAdd(accDot, rowDotSum);
}

// ---------------------------------------------------------------------------
// pt = e @ p  (M=16384, N=512, K=1024). 16x256 tile per wave: 16 independent
// WMMA chains reuse each A fragment, so e is read only twice from HBM
// (64 MB instead of 256 MB with narrow tiles). B read from pT [DIM][CLS]
// (L2-resident, K contiguous per lane). No LDS -> multi-wave/SIMD occupancy.
// ---------------------------------------------------------------------------
__global__ __launch_bounds__(32) void pt_gemm_kernel(const bf16_t* __restrict__ eb,
                                                     const bf16_t* __restrict__ pT,
                                                     float* __restrict__ pt) {
    const int lane = threadIdx.x;
    const int half = lane >> 4;
    const int l16  = lane & 15;
    const int m0   = (blockIdx.x >> 1) * 16;    // 1024 m-tiles
    const int n0   = (blockIdx.x & 1)  * 256;   // 2 n-groups of 256
    const int aRow = m0 + l16;

    v8f acc[16];
#pragma unroll
    for (int t = 0; t < 16; ++t) acc[t] = {};

    for (int kk = 0; kk < CLS; kk += 32) {
        const bf16_t* abase = eb + (size_t)aRow * CLS + kk + half * 16;
        v8bf lo = *(const v8bf*)(abase);
        v8bf hi = *(const v8bf*)(abase + 8);
        v16bf a;
#pragma unroll
        for (int i = 0; i < 8; ++i) { a[i] = lo[i]; a[i + 8] = hi[i]; }

        const bf16_t* bb = pT + (size_t)(n0 + l16) * CLS + kk + half * 16;
#pragma unroll
        for (int t = 0; t < 16; ++t) {
            v16bf b = *(const v16bf*)(bb + (size_t)(t * 16) * CLS);
            acc[t] = __builtin_amdgcn_wmma_f32_16x16x32_bf16(false, a, false, b,
                                                             (short)0, acc[t], false, false);
        }
    }
#pragma unroll
    for (int t = 0; t < 16; ++t) {
#pragma unroll
        for (int r = 0; r < 8; ++r) {
            const int m = m0 + half * 8 + r;
            pt[(size_t)m * DIM + n0 + t * 16 + l16] = acc[t][r];
        }
    }
}

// ---------------------------------------------------------------------------
// div_loss = |sum_k p_k|^2 - sum_k |p_k|^2 ; sim_loss = 1 - acc/B
// ---------------------------------------------------------------------------
__global__ __launch_bounds__(512) void finalize_kernel(const float* __restrict__ pf32,
                                                       const float* __restrict__ accDot,
                                                       float* __restrict__ outSim,
                                                       float* __restrict__ outDiv) {
    const int d = threadIdx.x;   // 512 threads, one per feature dim
    float s = 0.0f, tr = 0.0f;
    for (int k = 0; k < CLS; ++k) {
        float v = pf32[(size_t)k * DIM + d];
        s  += v;
        tr += v * v;
    }
    __shared__ float rs[512];
    __shared__ float rt[512];
    rs[d] = s * s;
    rt[d] = tr;
    __syncthreads();
    for (int o = 256; o > 0; o >>= 1) {
        if (d < o) { rs[d] += rs[d + o]; rt[d] += rt[d + o]; }
        __syncthreads();
    }
    if (d == 0) {
        *outDiv = rs[0] - rt[0];
        *outSim = 1.0f - accDot[0] * (1.0f / (float)BATCH);
    }
}

// ---------------------------------------------------------------------------
extern "C" void kernel_launch(void* const* d_in, const int* in_sizes, int n_in,
                              void* d_out, int out_size, void* d_ws, size_t ws_size,
                              hipStream_t stream) {
    const float* in_x = (const float*)d_in[0];   // [16384, 512]
    const float* in_p = (const float*)d_in[1];   // [1024, 512]
    const float* in_g = (const float*)d_in[2];   // [16384, 1024]

    float* out     = (float*)d_out;
    float* out_pt  = out;                                  // [16384,512]
    float* out_p   = out + (size_t)BATCH * DIM;            // [1024,512]
    float* out_sim = out + (size_t)BATCH * DIM + (size_t)CLS * DIM;
    float* out_div = out_sim + 1;

    char* ws = (char*)d_ws;
    bf16_t* xb     = (bf16_t*)(ws);                        // 16 MiB
    bf16_t* pb     = (bf16_t*)(ws + 16777216);             //  1 MiB
    bf16_t* pT     = (bf16_t*)(ws + 17825792);             //  1 MiB
    bf16_t* eb     = (bf16_t*)(ws + 18874368);             // 32 MiB
    float*  accDot = (float*) (ws + 52428800);             //  4 B

    zero_kernel<<<1, 32, 0, stream>>>(accDot);
    norm_x_kernel<<<BATCH, 256, 0, stream>>>(in_x, xb);
    norm_p_kernel<<<CLS, 256, 0, stream>>>(in_p, out_p, pb, pT);

    // 1024 blocks x 1 wave; LDS = 16*1028*4 bytes (padded sim tile)
    sim_softmax_kernel<<<BATCH / 16, 32, 16 * 1028 * 4, stream>>>(xb, pb, in_g, eb, accDot);

    // 16x256 tiles: 1024 m-tiles x 2 n-groups
    pt_gemm_kernel<<<(BATCH / 16) * (DIM / 256), 32, 0, stream>>>(eb, pT, out_pt);

    finalize_kernel<<<1, 512, 0, stream>>>(out_p, accDot, out_sim, out_div);
}
